// LaserDecoder_35914516529858
// MI455X (gfx1250) — compile-verified, matrix operand order checked by reading, not run on hardware
//
#include <hip/hip_runtime.h>

#define B_   32
#define S_   512
#define D_   512
#define H_   8
#define DK_  64
#define FF_  2048
#define ENC_ 1024
#define NL_  6
#define V_   8192
#define BS_  (B_*S_)

typedef __bf16 bf16;
typedef __attribute__((ext_vector_type(8)))  __bf16 v8bf;
typedef __attribute__((ext_vector_type(16))) __bf16 v16bf;
typedef __attribute__((ext_vector_type(8)))  float  v8f;

static __device__ __forceinline__ v16bf cat16(v8bf lo, v8bf hi) {
  return __builtin_shufflevector(lo, hi, 0,1,2,3,4,5,6,7,8,9,10,11,12,13,14,15);
}

// ---------------------------------------------------------------------------
// CDNA5 async global->LDS staging (ASYNCcnt path), guarded so the build can
// never regress: falls back to synchronous global_load+ds_store otherwise.
// Builtin signature (from compiler diagnostic): first arg is
// 'int __vector_size__(16) addrspace(1)*'; LDS side mirrored in addrspace(3).
// ---------------------------------------------------------------------------
#define AS1 __attribute__((address_space(1)))
#define AS3 __attribute__((address_space(3)))
typedef int i32x4 __attribute__((__vector_size__(16)));

#if defined(__gfx1250__) && __has_builtin(__builtin_amdgcn_global_load_async_to_lds_b128)
#define USE_ASYNC 1
#else
#define USE_ASYNC 0
#endif

static __device__ __forceinline__ AS3 i32x4* lds_cast(void* p) {
  // generic LDS address: low 32 bits are the LDS byte offset (aperture rule)
  return (AS3 i32x4*)(unsigned int)(unsigned long long)p;
}
static __device__ __forceinline__ AS1 i32x4* gbl_cast(const void* p) {
  return (AS1 i32x4*)(unsigned long long)p;
}
static __device__ __forceinline__ void wait_async0() {
#if USE_ASYNC
#if __has_builtin(__builtin_amdgcn_s_wait_asynccnt)
  __builtin_amdgcn_s_wait_asynccnt(0);
#else
  asm volatile("s_wait_asynccnt 0x0" ::: "memory");
#endif
#endif
}

// ---------------------------------------------------------------------------
// Weight preprocessing: fp32 [R][C] (row-major) -> bf16 [C][R] (transposed).
// Batched over blockIdx.z (stride R*C).
// ---------------------------------------------------------------------------
__global__ __launch_bounds__(256) void transpose_cvt(
    const float* __restrict__ in, bf16* __restrict__ out, int R, int C)
{
  __shared__ float tile[32][33];
  const float* src = in  + (size_t)blockIdx.z * R * C;
  bf16*        dst = out + (size_t)blockIdx.z * R * C;
  int r0 = blockIdx.y * 32, c0 = blockIdx.x * 32;
  int tx = threadIdx.x, ty = threadIdx.y;           // blockDim = (32, 8)
  #pragma unroll
  for (int i = 0; i < 32; i += 8)
    tile[ty + i][tx] = src[(size_t)(r0 + ty + i) * C + c0 + tx];
  __syncthreads();
  #pragma unroll
  for (int i = 0; i < 32; i += 8)
    dst[(size_t)(c0 + ty + i) * R + r0 + tx] = (bf16)tile[tx][ty + i];
}

// ---------------------------------------------------------------------------
// Embedding + positional + concat(encoder feats) -> bf16 [BS][D+ENC]
// ---------------------------------------------------------------------------
__global__ __launch_bounds__(256) void embed_concat(
    const float* __restrict__ src, const int* __restrict__ trg,
    const float* __restrict__ emb, const float* __restrict__ pos,
    bf16* __restrict__ Xc)
{
  int bs  = blockIdx.x;
  int s   = bs & (S_ - 1);
  int tok = trg[bs];
  const float scale = 22.62741699796952f;  // sqrt(512)
  for (int c = threadIdx.x; c < D_ + ENC_; c += 256) {
    float v;
    if (c < D_) v = emb[(size_t)tok * D_ + c] * scale + pos[(size_t)s * D_ + c];
    else        v = src[(size_t)bs * ENC_ + (c - D_)];
    Xc[(size_t)bs * (D_ + ENC_) + c] = (bf16)v;
  }
}

// ---------------------------------------------------------------------------
// Residual add + LayerNorm over D=512. One row per block, 256 threads.
// Writes fp32 (residual stream) and bf16 (next GEMM A operand).
// ---------------------------------------------------------------------------
__global__ __launch_bounds__(256) void ln_residual(
    float* __restrict__ y, const float* __restrict__ delta,
    const float* __restrict__ g, const float* __restrict__ bta,
    bf16* __restrict__ ybf)
{
  int row = blockIdx.x;
  size_t base = (size_t)row * D_;
  int t = threadIdx.x, lane = t & 31, w = t >> 5;
  float x0 = y[base + t]       + delta[base + t];
  float x1 = y[base + t + 256] + delta[base + t + 256];
  float s  = x0 + x1;
  float s2 = x0 * x0 + x1 * x1;
  #pragma unroll
  for (int off = 1; off < 32; off <<= 1) {
    s  += __shfl_xor(s,  off, 32);
    s2 += __shfl_xor(s2, off, 32);
  }
  __shared__ float ws1[8], ws2[8];
  if (lane == 0) { ws1[w] = s; ws2[w] = s2; }
  __syncthreads();
  float tot = 0.f, tot2 = 0.f;
  #pragma unroll
  for (int i = 0; i < 8; ++i) { tot += ws1[i]; tot2 += ws2[i]; }
  float mean = tot * (1.0f / D_);
  float var  = tot2 * (1.0f / D_) - mean * mean;
  float inv  = rsqrtf(var + 1e-5f);
  float o0 = (x0 - mean) * inv * g[t]       + bta[t];
  float o1 = (x1 - mean) * inv * g[t + 256] + bta[t + 256];
  y[base + t]         = o0;  ybf[base + t]         = (bf16)o0;
  y[base + t + 256]   = o1;  ybf[base + t + 256]   = (bf16)o1;
}

// ---------------------------------------------------------------------------
// bf16 WMMA GEMM: C[M,N] = A[M,K] (bf16, row-major) x W, with W given
// pre-transposed as Bt[N,K] (bf16, K contiguous). fp32 accumulation.
// Block tile 128x128x32, 8 waves (2x4), 64x32 per wave -> 8 WMMA / k-step.
// Double-buffered LDS; global->LDS via async (ASYNCcnt) when available.
// Epilogue: optional bias[N], optional ReLU, fp32 and/or bf16 outputs.
// M % 128 == 0, N % 128 == 0, K % 32 == 0 (true for all shapes here).
// ---------------------------------------------------------------------------
#define LDT 40   // padded LDS row stride (elems): 80B -> spreads banks
__global__ __launch_bounds__(256) void gemm_bf16_wmma(
    const bf16* __restrict__ A, const bf16* __restrict__ Bt,
    const float* __restrict__ bias, float* __restrict__ outF,
    bf16* __restrict__ outH, int M, int N, int K, int relu)
{
  __shared__ __align__(16) bf16 sA[2][128 * LDT];
  __shared__ __align__(16) bf16 sB[2][128 * LDT];
  const int t    = threadIdx.x;
  const int lane = t & 31, w = t >> 5;
  const int wm   = w >> 2, wn = w & 3;
  const int lr   = lane & 15, half = lane >> 4;
  const int m0   = blockIdx.y * 128, n0 = blockIdx.x * 128;

  // per-thread staging coordinates: 2 x v8bf for A and for B per k-step
  const int r0s = t >> 2,           kc0 = (t & 3) << 3;
  const int r1s = (t + 256) >> 2,   kc1 = ((t + 256) & 3) << 3;

  auto stage = [&](int buf, int k0) {
#if USE_ASYNC
    __builtin_amdgcn_global_load_async_to_lds_b128(
        gbl_cast(A  + (size_t)(m0 + r0s) * K + k0 + kc0),
        lds_cast(&sA[buf][r0s * LDT + kc0]), 0, 0);
    __builtin_amdgcn_global_load_async_to_lds_b128(
        gbl_cast(A  + (size_t)(m0 + r1s) * K + k0 + kc1),
        lds_cast(&sA[buf][r1s * LDT + kc1]), 0, 0);
    __builtin_amdgcn_global_load_async_to_lds_b128(
        gbl_cast(Bt + (size_t)(n0 + r0s) * K + k0 + kc0),
        lds_cast(&sB[buf][r0s * LDT + kc0]), 0, 0);
    __builtin_amdgcn_global_load_async_to_lds_b128(
        gbl_cast(Bt + (size_t)(n0 + r1s) * K + k0 + kc1),
        lds_cast(&sB[buf][r1s * LDT + kc1]), 0, 0);
#else
    *(v8bf*)(&sA[buf][r0s * LDT + kc0]) =
        *(const v8bf*)(A  + (size_t)(m0 + r0s) * K + k0 + kc0);
    *(v8bf*)(&sA[buf][r1s * LDT + kc1]) =
        *(const v8bf*)(A  + (size_t)(m0 + r1s) * K + k0 + kc1);
    *(v8bf*)(&sB[buf][r0s * LDT + kc0]) =
        *(const v8bf*)(Bt + (size_t)(n0 + r0s) * K + k0 + kc0);
    *(v8bf*)(&sB[buf][r1s * LDT + kc1]) =
        *(const v8bf*)(Bt + (size_t)(n0 + r1s) * K + k0 + kc1);
    if (k0 + 32 < K) {  // global_prefetch_b8 of next k-panel (fallback only)
      __builtin_prefetch((const void*)(A  + (size_t)(m0 + (t >> 1)) * K + k0 + 32), 0, 0);
      __builtin_prefetch((const void*)(Bt + (size_t)(n0 + (t >> 1)) * K + k0 + 32), 0, 0);
    }
#endif
  };

  const v8f z8 = {0.f,0.f,0.f,0.f,0.f,0.f,0.f,0.f};
  v8f acc[4][2];
  #pragma unroll
  for (int a0 = 0; a0 < 4; ++a0)
    #pragma unroll
    for (int b0 = 0; b0 < 2; ++b0) acc[a0][b0] = z8;

  const int nk = K >> 5;
  stage(0, 0);
  wait_async0();
  __syncthreads();

  for (int kt = 0; kt < nk; ++kt) {
    const int buf = kt & 1;
    if (kt + 1 < nk) stage(buf ^ 1, (kt + 1) << 5);  // overlap with WMMAs

    const bf16* pa = &sA[buf][0];
    const bf16* pb = &sB[buf][0];
    v16bf af[4], bfr[2];
    #pragma unroll
    for (int tm = 0; tm < 4; ++tm) {
      const bf16* p = pa + (wm * 64 + tm * 16 + lr) * LDT;
      af[tm] = cat16(*(const v8bf*)(p + half * 8),
                     *(const v8bf*)(p + 16 + half * 8));
    }
    #pragma unroll
    for (int tn = 0; tn < 2; ++tn) {
      const bf16* p = pb + (wn * 32 + tn * 16 + lr) * LDT;
      bfr[tn] = cat16(*(const v8bf*)(p + half * 8),
                      *(const v8bf*)(p + 16 + half * 8));
    }
    #pragma unroll
    for (int tm = 0; tm < 4; ++tm)
      #pragma unroll
      for (int tn = 0; tn < 2; ++tn)
        acc[tm][tn] = __builtin_amdgcn_wmma_f32_16x16x32_bf16(
            false, af[tm], false, bfr[tn], (short)0, acc[tm][tn], false, false);

    wait_async0();   // tile kt+1 landed while WMMAs ran
    __syncthreads();
  }

  // epilogue: C/D layout -> lane holds (row = j + 8*half, col = lr) per tile
  #pragma unroll
  for (int tn = 0; tn < 2; ++tn) {
    int col = n0 + wn * 32 + tn * 16 + lr;
    float bv = bias ? bias[col] : 0.0f;
    #pragma unroll
    for (int tm = 0; tm < 4; ++tm) {
      int rbase = m0 + wm * 64 + tm * 16 + 8 * half;
      #pragma unroll
      for (int j = 0; j < 8; ++j) {
        float v = acc[tm][tn][j] + bv;
        if (relu) v = fmaxf(v, 0.0f);
        size_t idx = (size_t)(rbase + j) * N + col;
        if (outF) outF[idx] = v;
        if (outH) outH[idx] = (bf16)v;
      }
    }
  }
}

// ---------------------------------------------------------------------------
// Causal flash attention, one (b, h, 64-q-row block) per workgroup.
// 4 waves x 16 q-rows. 32-key tiles: QK^T via 2 WMMA per 16x16 score tile,
// online softmax in fp32, P routed through LDS (C-layout -> A-layout),
// O += P @ V via 4 WMMA. Q/K/V/O are bf16 [B*S][D], head h at column h*64.
// ---------------------------------------------------------------------------
__global__ __launch_bounds__(128) void flash_attn(
    const bf16* __restrict__ Q, const bf16* __restrict__ Km,
    const bf16* __restrict__ Vm, bf16* __restrict__ O)
{
  __shared__ __align__(16) bf16 sK[32 * 72];    // [key][dk], stride 72
  __shared__ __align__(16) bf16 sVT[64 * 40];   // [dk][key], stride 40
  __shared__ __align__(16) bf16 sP[4 * 16 * 40];// per-wave [qrow][key]
  const int qblk = blockIdx.x, h = blockIdx.y, b = blockIdx.z;
  const int t = threadIdx.x, lane = t & 31, w = t >> 5;
  const int lr = lane & 15, half = lane >> 4;
  const size_t headOfs = ((size_t)b * S_) * D_ + h * DK_;
  const bf16* Qb = Q  + headOfs;
  const bf16* Kb = Km + headOfs;
  const bf16* Vb = Vm + headOfs;
  bf16* sPw = sP + w * (16 * 40);
  const int qrb = qblk * 64 + w * 16;

  // Q fragments (16 rows x dk=64): A-layout loads straight from global
  const bf16* qp = Qb + (size_t)(qrb + lr) * D_;
  v16bf aQ0 = cat16(*(const v8bf*)(qp + half * 8),
                    *(const v8bf*)(qp + 16 + half * 8));
  v16bf aQ1 = cat16(*(const v8bf*)(qp + 32 + half * 8),
                    *(const v8bf*)(qp + 48 + half * 8));

  const v8f z8 = {0.f,0.f,0.f,0.f,0.f,0.f,0.f,0.f};
  v8f o[4];
  #pragma unroll
  for (int d = 0; d < 4; ++d) o[d] = z8;
  float m[8], l[8];
  #pragma unroll
  for (int j = 0; j < 8; ++j) { m[j] = -1e30f; l[j] = 0.f; }

  const int nkt = qblk * 2 + 2;                 // causal bound for this block
  for (int kt = 0; kt < nkt; ++kt) {
    const int kb = kt * 32;
    // cooperative staging: K as-is, V transposed
    #pragma unroll
    for (int lo2 = 0; lo2 < 2; ++lo2) {
      int u = t + lo2 * 128;
      int key = u >> 3, seg = u & 7;
      v8bf kv = *(const v8bf*)(Kb + (size_t)(kb + key) * D_ + seg * 8);
      *(v8bf*)(sK + key * 72 + seg * 8) = kv;
      v8bf vv = *(const v8bf*)(Vb + (size_t)(kb + key) * D_ + seg * 8);
      #pragma unroll
      for (int d2 = 0; d2 < 8; ++d2)
        sVT[(seg * 8 + d2) * 40 + key] = vv[d2];
    }
    __syncthreads();

    if (kb <= qrb + 15) {                       // wave-uniform causal skip
      v8f sc[2]; sc[0] = z8; sc[1] = z8;
      #pragma unroll
      for (int tn = 0; tn < 2; ++tn) {
        const bf16* kp = sK + (tn * 16 + lr) * 72;
        v16bf b0 = cat16(*(const v8bf*)(kp + half * 8),
                         *(const v8bf*)(kp + 16 + half * 8));
        v16bf b1 = cat16(*(const v8bf*)(kp + 32 + half * 8),
                         *(const v8bf*)(kp + 48 + half * 8));
        sc[tn] = __builtin_amdgcn_wmma_f32_16x16x32_bf16(
            false, aQ0, false, b0, (short)0, sc[tn], false, false);
        sc[tn] = __builtin_amdgcn_wmma_f32_16x16x32_bf16(
            false, aQ1, false, b1, (short)0, sc[tn], false, false);
      }
      const float rs = 0.125f;                  // 1/sqrt(64)
      #pragma unroll
      for (int j = 0; j < 8; ++j) {
        int q = qrb + j + 8 * half;
        float s0 = sc[0][j] * rs;
        float s1 = sc[1][j] * rs;
        if (kb + lr > q)      s0 = -1e30f;      // causal mask
        if (kb + 16 + lr > q) s1 = -1e30f;
        float mm = fmaxf(s0, s1);
        #pragma unroll
        for (int off = 1; off < 16; off <<= 1)
          mm = fmaxf(mm, __shfl_xor(mm, off, 32));
        float mn   = fmaxf(m[j], mm);
        float corr = __expf(m[j] - mn);
        m[j] = mn;
        float p0 = __expf(s0 - mn);
        float p1 = __expf(s1 - mn);
        float ls = p0 + p1;
        #pragma unroll
        for (int off = 1; off < 16; off <<= 1)
          ls += __shfl_xor(ls, off, 32);
        l[j] = l[j] * corr + ls;
        #pragma unroll
        for (int d = 0; d < 4; ++d) o[d][j] = o[d][j] * corr;
        sPw[(j + 8 * half) * 40 + lr]      = (bf16)p0;
        sPw[(j + 8 * half) * 40 + 16 + lr] = (bf16)p1;
      }
      // P back in A-layout (same-wave DS ops are in-order)
      v16bf pA = cat16(*(const v8bf*)(sPw + lr * 40 + half * 8),
                       *(const v8bf*)(sPw + lr * 40 + 16 + half * 8));
      #pragma unroll
      for (int d = 0; d < 4; ++d) {
        const bf16* vp = sVT + (d * 16 + lr) * 40;
        v16bf vB = cat16(*(const v8bf*)(vp + half * 8),
                         *(const v8bf*)(vp + 16 + half * 8));
        o[d] = __builtin_amdgcn_wmma_f32_16x16x32_bf16(
            false, pA, false, vB, (short)0, o[d], false, false);
      }
    }
    __syncthreads();
  }

  #pragma unroll
  for (int d = 0; d < 4; ++d)
    #pragma unroll
    for (int j = 0; j < 8; ++j) {
      int row = qrb + j + 8 * half;
      int col = h * DK_ + d * 16 + lr;
      O[((size_t)b * S_ + row) * D_ + col] = (bf16)(o[d][j] / l[j]);
    }
}

// ---------------------------------------------------------------------------
// Host orchestration
// ---------------------------------------------------------------------------
extern "C" void kernel_launch(void* const* d_in, const int* in_sizes, int n_in,
                              void* d_out, int out_size, void* d_ws, size_t ws_size,
                              hipStream_t stream)
{
  (void)in_sizes; (void)n_in; (void)out_size; (void)ws_size;
  const float* src   = (const float*)d_in[0];
  const int*   trg   = (const int*)  d_in[1];
  const float* emb   = (const float*)d_in[2];
  const float* pos   = (const float*)d_in[3];
  const float* projW = (const float*)d_in[4];
  const float* projB = (const float*)d_in[5];
  const float* Wq    = (const float*)d_in[6];
  const float* Wk    = (const float*)d_in[7];
  const float* Wv    = (const float*)d_in[8];
  const float* Wo    = (const float*)d_in[9];
  const float* ln1g  = (const float*)d_in[10];
  const float* ln1b  = (const float*)d_in[11];
  const float* ln2g  = (const float*)d_in[12];
  const float* ln2b  = (const float*)d_in[13];
  const float* ffW1  = (const float*)d_in[14];
  const float* ffb1  = (const float*)d_in[15];
  const float* ffW2  = (const float*)d_in[16];
  const float* ffb2  = (const float*)d_in[17];
  const float* outW  = (const float*)d_in[18];
  const float* outB  = (const float*)d_in[19];
  float* out = (float*)d_out;

  // workspace arena (~199 MB)
  char* ws = (char*)d_ws;
  size_t off = 0;
  auto alloc = [&](size_t bytes) -> void* {
    void* p = ws + off;
    off = (off + bytes + 255) & ~(size_t)255;
    return p;
  };
  bf16* projT = (bf16*)alloc((size_t)(D_+ENC_) * D_ * 2);
  bf16* WqT   = (bf16*)alloc((size_t)NL_ * D_ * D_ * 2);
  bf16* WkT   = (bf16*)alloc((size_t)NL_ * D_ * D_ * 2);
  bf16* WvT   = (bf16*)alloc((size_t)NL_ * D_ * D_ * 2);
  bf16* WoT   = (bf16*)alloc((size_t)NL_ * D_ * D_ * 2);
  bf16* ff1T  = (bf16*)alloc((size_t)NL_ * D_ * FF_ * 2);
  bf16* ff2T  = (bf16*)alloc((size_t)NL_ * D_ * FF_ * 2);
  bf16* outT  = (bf16*)alloc((size_t)D_ * V_ * 2);
  float* y32  = (float*)alloc((size_t)BS_ * D_ * 4);
  bf16* ybf   = (bf16*) alloc((size_t)BS_ * D_ * 2);
  float* delta= (float*)alloc((size_t)BS_ * D_ * 4);
  char*  big  = (char*) alloc((size_t)BS_ * FF_ * 2);  // reused region
  bf16* Xc    = (bf16*)big;                            // [BS][1536]
  bf16* Qb    = (bf16*)big;                            // [BS][512]
  bf16* Kbuf  = Qb   + (size_t)BS_ * D_;
  bf16* Vbuf  = Kbuf + (size_t)BS_ * D_;
  bf16* attnb = Vbuf + (size_t)BS_ * D_;
  bf16* ffh   = (bf16*)big;                            // [BS][2048]

  dim3 tb(32, 8);
  transpose_cvt<<<dim3(D_/32, (D_+ENC_)/32, 1),  tb, 0, stream>>>(projW, projT, D_+ENC_, D_);
  transpose_cvt<<<dim3(D_/32,  D_/32,  NL_),     tb, 0, stream>>>(Wq,  WqT,  D_,  D_);
  transpose_cvt<<<dim3(D_/32,  D_/32,  NL_),     tb, 0, stream>>>(Wk,  WkT,  D_,  D_);
  transpose_cvt<<<dim3(D_/32,  D_/32,  NL_),     tb, 0, stream>>>(Wv,  WvT,  D_,  D_);
  transpose_cvt<<<dim3(D_/32,  D_/32,  NL_),     tb, 0, stream>>>(Wo,  WoT,  D_,  D_);
  transpose_cvt<<<dim3(FF_/32, D_/32,  NL_),     tb, 0, stream>>>(ffW1, ff1T, D_,  FF_);
  transpose_cvt<<<dim3(D_/32,  FF_/32, NL_),     tb, 0, stream>>>(ffW2, ff2T, FF_, D_);
  transpose_cvt<<<dim3(V_/32,  D_/32,  1),       tb, 0, stream>>>(outW, outT, D_,  V_);

  embed_concat<<<BS_, 256, 0, stream>>>(src, trg, emb, pos, Xc);

  auto gemm = [&](const bf16* A, const bf16* Bt, const float* bias,
                  float* oF, bf16* oH, int M, int N, int K, int relu) {
    gemm_bf16_wmma<<<dim3(N/128, M/128), 256, 0, stream>>>(A, Bt, bias, oF, oH, M, N, K, relu);
  };

  // input projection: y = Xc @ proj_W + proj_b
  gemm(Xc, projT, projB, y32, ybf, BS_, D_, D_+ENC_, 0);

  for (int i = 0; i < NL_; ++i) {
    gemm(ybf, WqT + (size_t)i*D_*D_, nullptr, nullptr, Qb,   BS_, D_, D_, 0);
    gemm(ybf, WkT + (size_t)i*D_*D_, nullptr, nullptr, Kbuf, BS_, D_, D_, 0);
    gemm(ybf, WvT + (size_t)i*D_*D_, nullptr, nullptr, Vbuf, BS_, D_, D_, 0);
    flash_attn<<<dim3(S_/64, H_, B_), 128, 0, stream>>>(Qb, Kbuf, Vbuf, attnb);
    gemm(attnb, WoT + (size_t)i*D_*D_, nullptr, delta, nullptr, BS_, D_, D_, 0);
    ln_residual<<<BS_, 256, 0, stream>>>(y32, delta, ln1g + i*D_, ln1b + i*D_, ybf);
    gemm(ybf, ff1T + (size_t)i*D_*FF_, ffb1 + (size_t)i*FF_, nullptr, ffh, BS_, FF_, D_, 1);
    gemm(ffh, ff2T + (size_t)i*D_*FF_, ffb2 + (size_t)i*D_,  delta, nullptr, BS_, D_, FF_, 0);
    ln_residual<<<BS_, 256, 0, stream>>>(y32, delta, ln2g + i*D_, ln2b + i*D_, ybf);
  }

  // logits
  gemm(ybf, outT, outB, out, nullptr, BS_, V_, D_, 0);
}
